// Pyramid_26800595927669
// MI455X (gfx1250) — matrix operand order repaired and checked by reading, not compiled, not run on hardware
//
#include <hip/hip_runtime.h>
#include <hip/hip_bf16.h>
#include <math.h>
#include <cstdint>

// ---------------------------------------------------------------------------
// Pyramid (SPINN tree-LSTM) for MI455X / gfx1250.
// wave32 WMMA bf16 GEMMs + double-buffered async global->LDS staging:
//   issue(tile k+1) -> s_wait_asynccnt <= inflight(k+1) -> barrier -> wmma(k)
// Weights pre-packed per launch into WMMA fragment order (bf16); states kept
// as f32 master + bf16 shadow so all hot-loop staging is pure async copy.
// ---------------------------------------------------------------------------

typedef __attribute__((ext_vector_type(16))) __bf16 v16bf;
typedef __attribute__((ext_vector_type(8)))  float  v8f;

#define NB    128
#define SSEQ  64
#define DST   512     // state dim = 2H
#define HH    256
#define NGATE 1280    // 5H
#define SELD  128
#define MLPDD 1024

__device__ __forceinline__ float sigm(float x) { return 1.0f / (1.0f + __expf(-x)); }

__device__ __forceinline__ v8f wmma_bf16(v16bf a, v16bf b, v8f c) {
  return __builtin_amdgcn_wmma_f32_16x16x32_bf16(false, a, false, b, (short)0, c,
                                                 false, false);
}

// CDNA5 async global->LDS copy (tracked by ASYNCcnt).
__device__ __forceinline__ void async_b128(uint32_t lds, const void* g) {
  asm volatile("global_load_async_to_lds_b128 %0, %1, off" :: "v"(lds), "v"(g) : "memory");
}
__device__ __forceinline__ void async_b64(uint32_t lds, const void* g) {
  asm volatile("global_load_async_to_lds_b64 %0, %1, off" :: "v"(lds), "v"(g) : "memory");
}
__device__ __forceinline__ void wait_async0() { asm volatile("s_wait_asynccnt 0x0" ::: "memory"); }
__device__ __forceinline__ void wait_async2() { asm volatile("s_wait_asynccnt 0x2" ::: "memory"); }
__device__ __forceinline__ void wait_async3() { asm volatile("s_wait_asynccnt 0x3" ::: "memory"); }
__device__ __forceinline__ void wait_async4() { asm volatile("s_wait_asynccnt 0x4" ::: "memory"); }
__device__ __forceinline__ uint32_t lds_addr(const void* p) {
  return (uint32_t)(uintptr_t)p;   // generic shared ptr: low 32 bits = LDS offset
}

// CDNA5 wave32 fragment swizzle (16-bit operands, 16x16x32):
// A (16 rows x 32 K): lane = (row&15) | (((k>>3)&1)<<4), i = (k&7) | ((k&16)>>1)
//   -> 8 consecutive K (one octet) are contiguous 16B in LDS: async-copyable.
// B (32 K x 16 cols): lane = (col&15) | (((k>>3)&1)<<4), i = (k&7) | ((k&16)>>1) (same as A)
__device__ __forceinline__ int a_lane(int row, int k) { return (row & 15) | (((k >> 3) & 1) << 4); }
__device__ __forceinline__ int a_idx (int k)          { return (k & 7) | ((k & 16) >> 1); }

// ---------------------------------------------------------------------------
// Weight packing (once per launch): f32 [K,N] -> bf16 fragment-ordered tiles.
// comp_W: [kb][jb][gate(5)][jt(2)][lane(32)][i(16)]  (per (kb,jb): 10240 B)
// generic: [kb][nb][nt(2)][lane(32)][i(16)]          (per (kb,nb):  2048 B)
// ---------------------------------------------------------------------------
__global__ void pack_compW_kernel(const float* __restrict__ W, __bf16* __restrict__ out)
{
  const int idx = blockIdx.x * 256 + threadIdx.x;
  if (idx >= DST * NGATE) return;
  const int k = idx / NGATE, n = idx - k * NGATE;
  const int kb = k >> 5, kk = k & 31;
  const int gg = n / HH, j = n - gg * HH;
  const int jb = j >> 5, nl = j & 31;
  const size_t dst = ((((size_t)(kb * 8 + jb) * 5 + gg) * 2 + (nl >> 4)) * 32
                      + ((nl & 15) | (((kk >> 3) & 1) << 4))) * 16 + ((kk & 7) | ((kk & 16) >> 1));
  out[dst] = (__bf16)W[idx];
}

__global__ void pack_w_kernel(const float* __restrict__ W, __bf16* __restrict__ out,
                              int K, int N)
{
  const int idx = blockIdx.x * 256 + threadIdx.x;
  if (idx >= K * N) return;
  const int k = idx / N, n = idx - k * N;
  const int kb = k >> 5, kk = k & 31;
  const size_t dst = (((size_t)kb * (N >> 5) + (n >> 5)) * 2 + ((n >> 4) & 1)) * 512
                     + (size_t)((n & 15) | (((kk >> 3) & 1) << 4)) * 16 + ((kk & 7) | ((kk & 16) >> 1));
  out[dst] = (__bf16)W[idx];
}

// ---------------------------------------------------------------------------
// Embedding gather: f32 master + bf16 shadow.
// ---------------------------------------------------------------------------
__global__ void embed_kernel(const int* __restrict__ sent,
                             const float* __restrict__ table,
                             float* __restrict__ st, __bf16* __restrict__ stbf)
{
  const int row = blockIdx.x;          // b*64 + s
  const int d4  = threadIdx.x;         // 0..127
  const int tok = sent[row];
  const size_t off = (size_t)row * DST + d4 * 4;
  const float4 v = *reinterpret_cast<const float4*>(table + (size_t)tok * DST + d4 * 4);
  *reinterpret_cast<float4*>(st + off) = v;
  __bf16* d = stbf + off;
  d[0] = (__bf16)v.x; d[1] = (__bf16)v.y; d[2] = (__bf16)v.z; d[3] = (__bf16)v.w;
}

// ---------------------------------------------------------------------------
// Tree-LSTM composition: gates = [h_l||h_r] @ comp_W + b, fused LSTM cell.
// 256 thr / 8 waves; tile 64 M x 32 hidden units; 5 WMMA accs per wave.
// Double-buffered async staging: 4 async issues / tile / wave.
// ---------------------------------------------------------------------------
__global__ __launch_bounds__(256) void gates_kernel(
    const float* __restrict__ statesF, const __bf16* __restrict__ statesBF,
    const __bf16* __restrict__ pwC, const float* __restrict__ compb,
    float* __restrict__ compbuf, int L)
{
  const int M = NB * L;
  __shared__ alignas(32) __bf16 Asw[2][4][32][16];      //  8 KB (2 buffers)
  __shared__ alignas(32) __bf16 Bsw[2][5][2][32][16];   // 20 KB (2 buffers)

  const int t    = threadIdx.x;
  const int lane = t & 31;
  const int wave = t >> 5;
  const int wm   = wave & 3;
  const int wj   = wave >> 2;
  const int m0   = blockIdx.x * 64;
  const int j0   = blockIdx.y * 32;

  // Per-thread async-copy assignment (row clamped -> unconditional issue,
  // keeps per-wave ASYNCcnt exactly 4 per tile).
  const int rA = t >> 2;               // tile row 0..63
  const int oA = t & 3;                // K octet 0..3
  const int mA = (m0 + rA < M) ? (m0 + rA) : (M - 1);
  const int bA = mA / L, pA = mA - bA * L;
  const uint32_t ldsAoff = lds_addr(&Asw[0][0][0][0]) +
      (uint32_t)(((rA >> 4) * 512 + ((rA & 15) | ((oA & 1) << 4)) * 16 + (oA >> 1) * 8) * 2);
  const uint32_t ldsB = lds_addr(&Bsw[0][0][0][0][0]);
  const char* pwBase = (const char*)(pwC + (size_t)(j0 >> 5) * 5120);

  auto issue = [&](int kb, int buf) {
    const int kg  = (kb << 5) + oA * 8;
    const int pos = pA + (kg >> 8);              // left h / right h
    async_b128(ldsAoff + buf * 4096,
               statesBF + ((size_t)(bA * SSEQ + pos)) * DST + (kg & 255));
    const char* pb = pwBase + (size_t)kb * (8 * 10240);
    const uint32_t lb = ldsB + buf * 10240;
    async_b128(lb + t * 16,        pb + t * 16);
    async_b128(lb + 4096 + t * 16, pb + 4096 + t * 16);
    async_b64 (lb + 8192 + t * 8,  pb + 8192 + t * 8);
    __builtin_prefetch(pb + 8 * 10240, 0, 1);
  };

  v8f acc[5] = {};
  const int NKB = DST / 32;            // 16 K-blocks

  issue(0, 0);
  for (int kb = 0; kb < NKB; ++kb) {
    const int cur = kb & 1;
    __syncthreads();                   // (a) prior consumers of buf cur^1 done
    const bool more = (kb + 1 < NKB);
    if (more) { issue(kb + 1, cur ^ 1); wait_async4(); }   // tile kb resident
    else      { wait_async0(); }
    __syncthreads();                   // (b) tile kb visible to all waves

    const v16bf a = *reinterpret_cast<const v16bf*>(&Asw[cur][wm][lane][0]);
    #pragma unroll
    for (int gg = 0; gg < 5; ++gg) {
      const v16bf bb = *reinterpret_cast<const v16bf*>(&Bsw[cur][gg][wj][lane][0]);
      acc[gg] = wmma_bf16(a, bb, acc[gg]);
    }
  }

  // Fused tree-LSTM cell epilogue (f32 precision).
  const int jloc  = j0 + wj * 16 + (lane & 15);
  const int rowhi = (lane >> 4) * 8;
  const float bi  = compb[jloc];
  const float bfl = compb[HH + jloc];
  const float bfr = compb[2 * HH + jloc];
  const float bu  = compb[3 * HH + jloc];
  const float bo  = compb[4 * HH + jloc];
  #pragma unroll
  for (int r = 0; r < 8; ++r) {
    const int m = m0 + wm * 16 + rowhi + r;
    if (m >= M) continue;
    const int b = m / L, p = m - b * L;
    const float gi  = acc[0][r] + bi;
    const float gfl = acc[1][r] + bfl;
    const float gfr = acc[2][r] + bfr;
    const float gu  = acc[3][r] + bu;
    const float go  = acc[4][r] + bo;
    const float cl = statesF[((size_t)(b * SSEQ + p)) * DST + HH + jloc];
    const float cr = statesF[((size_t)(b * SSEQ + p + 1)) * DST + HH + jloc];
    const float c  = sigm(gfl) * cl + sigm(gfr) * cr + sigm(gi) * tanhf(gu);
    const float h  = sigm(go) * tanhf(c);
    float* dst = compbuf + ((size_t)(b * SSEQ + p)) * DST;
    dst[jloc]      = h;
    dst[HH + jloc] = c;
  }
}

// ---------------------------------------------------------------------------
// Selection scorer: logit = tanh([left||right] @ W1 + b1) @ W2 + b2.
// 256 thr / 8 waves; 16 M-rows x 128 N; K = 1024; double-buffered async.
// Waves 0-1 issue 3 asyncs/tile (A+2B), waves 2-7 issue 2 (wave-uniform).
// ---------------------------------------------------------------------------
__global__ __launch_bounds__(256) void sel_kernel(
    const __bf16* __restrict__ statesBF, const __bf16* __restrict__ pwS,
    const float* __restrict__ b1, const float* __restrict__ W2,
    const float* __restrict__ b2, float* __restrict__ logits, int L)
{
  const int M = NB * L;
  __shared__ alignas(32) __bf16 Asw[2][32][16];        //  2 KB
  __shared__ alignas(32) __bf16 Bsw[2][8][32][16];     // 16 KB
  __shared__ float part[16][SELD];                     //  8 KB

  const int t    = threadIdx.x;
  const int lane = t & 31;
  const int wave = t >> 5;
  const int m0   = blockIdx.x * 16;
  const bool hasA = (t < 64);                          // wave-uniform

  const int rA = t >> 2, oA = t & 3;
  const int mA = (m0 + rA < M) ? (m0 + rA) : (M - 1);  // clamped
  const int bA = mA / L, pA = mA - bA * L;
  const uint32_t ldsAoff = lds_addr(&Asw[0][0][0]) +
      (uint32_t)((((rA & 15) | ((oA & 1) << 4)) * 16 + (oA >> 1) * 8) * 2);
  const uint32_t ldsB = lds_addr(&Bsw[0][0][0][0]);

  auto issue = [&](int kb, int buf) {
    if (hasA) {
      const int kg  = (kb << 5) + oA * 8;
      const int pos = pA + (kg >> 9);                  // left state / right
      async_b128(ldsAoff + buf * 1024,
                 statesBF + ((size_t)(bA * SSEQ + pos)) * DST + (kg & 511));
    }
    const char* pb = (const char*)(pwS + (size_t)kb * 4096);
    const uint32_t lb = ldsB + buf * 8192;
    async_b128(lb + t * 16,        pb + t * 16);
    async_b128(lb + 4096 + t * 16, pb + 4096 + t * 16);
  };

  v8f acc = {};
  const int NKB = (2 * DST) / 32;      // 32 K-blocks

  issue(0, 0);
  for (int kb = 0; kb < NKB; ++kb) {
    const int cur = kb & 1;
    __syncthreads();
    const bool more = (kb + 1 < NKB);
    if (more) {
      issue(kb + 1, cur ^ 1);
      if (hasA) wait_async3(); else wait_async2();
    } else {
      wait_async0();
    }
    __syncthreads();

    const v16bf a  = *reinterpret_cast<const v16bf*>(&Asw[cur][lane][0]);
    const v16bf bb = *reinterpret_cast<const v16bf*>(&Bsw[cur][wave][lane][0]);
    acc = wmma_bf16(a, bb, acc);
  }

  const int n     = wave * 16 + (lane & 15);
  const int rowhi = (lane >> 4) * 8;
  const float w2  = W2[n];
  const float bb1 = b1[n];
  #pragma unroll
  for (int r = 0; r < 8; ++r)
    part[rowhi + r][n] = tanhf(acc[r] + bb1) * w2;
  __syncthreads();
  if (t < 16) {
    const int m = m0 + t;
    if (m < M) {
      float s = b2[0];
      for (int n2 = 0; n2 < SELD; ++n2) s += part[t][n2];
      const int b = m / L, p = m - b * L;
      logits[b * SSEQ + p] = s;
    }
  }
}

// ---------------------------------------------------------------------------
// Softmax over positions + cumsum mixing; writes f32 master + bf16 shadow.
// ---------------------------------------------------------------------------
__global__ __launch_bounds__(256) void update_kernel(
    const float* __restrict__ states, const float* __restrict__ compbuf,
    const float* __restrict__ logits, float* __restrict__ out,
    __bf16* __restrict__ outbf, int L)
{
  __shared__ float pr[SSEQ], cl[SSEQ], cr[SSEQ];
  const int b = blockIdx.x;
  const int t = threadIdx.x;
  if (t < L) pr[t] = logits[b * SSEQ + t];
  __syncthreads();
  if (t == 0) {
    float mx = -1e30f;
    for (int p = 0; p < L; ++p) mx = fmaxf(mx, pr[p]);
    float sum = 0.f;
    for (int p = 0; p < L; ++p) { const float e = __expf(pr[p] - mx); pr[p] = e; sum += e; }
    const float inv = 1.f / sum;
    float cs = 0.f;
    for (int p = 0; p < L; ++p) {
      const float prob = pr[p] * inv;
      pr[p] = prob;        // select
      cr[p] = cs;          // copy_right = cumsum_before
      cs   += prob;
      cl[p] = cs;
    }
    for (int p = 0; p < L; ++p) cl[p] = cs - cl[p];   // copy_left
  }
  __syncthreads();
  const int total = L * (DST / 4);
  for (int idx = t; idx < total; idx += 256) {
    const int p  = idx >> 7;
    const int d4 = (idx & 127) * 4;
    const size_t lo = ((size_t)(b * SSEQ + p)) * DST + d4;
    const float4 lv = *reinterpret_cast<const float4*>(states + lo);
    const float4 rv = *reinterpret_cast<const float4*>(states + lo + DST);
    const float4 cv = *reinterpret_cast<const float4*>(compbuf + lo);
    const float a_ = cl[p], b_ = cr[p], s_ = pr[p];
    float4 o;
    o.x = a_ * lv.x + b_ * rv.x + s_ * cv.x;
    o.y = a_ * lv.y + b_ * rv.y + s_ * cv.y;
    o.z = a_ * lv.z + b_ * rv.z + s_ * cv.z;
    o.w = a_ * lv.w + b_ * rv.w + s_ * cv.w;
    *reinterpret_cast<float4*>(out + lo) = o;
    __bf16* d = outbf + lo;
    d[0] = (__bf16)o.x; d[1] = (__bf16)o.y; d[2] = (__bf16)o.z; d[3] = (__bf16)o.w;
  }
}

// ---------------------------------------------------------------------------
// Generic WMMA GEMM: out = act(A[M,K](f32,lda) @ Wpacked[K,N] + bias).
// A staged via cvt path (f32 source); B via async copy of packed bf16.
// ---------------------------------------------------------------------------
__global__ __launch_bounds__(256) void mlp_gemm_kernel(
    const float* __restrict__ A, int lda, int M, int K, int N,
    const __bf16* __restrict__ pw, const float* __restrict__ bias,
    float* __restrict__ out, int relu)
{
  __shared__ alignas(32) __bf16 Asw[4][32][16];
  __shared__ alignas(32) __bf16 Bsw[2][32][16];

  const int t    = threadIdx.x;
  const int lane = t & 31;
  const int wave = t >> 5;
  const int wm   = wave & 3;
  const int wn   = wave >> 2;
  const int m0   = blockIdx.x * 64;
  const int nb0  = blockIdx.y;                // 32-col block
  const uint32_t ldsB = lds_addr(&Bsw[0][0][0]);

  v8f acc = {};

  for (int k0 = 0, kb = 0; k0 < K; k0 += 32, ++kb) {
    __syncthreads();
    #pragma unroll
    for (int q = 0; q < 2; ++q) {
      const int f  = t * 2 + q;
      const int r  = f >> 3;
      const int kk = (f & 7) * 4;
      const int m  = m0 + r;
      float4 v = make_float4(0.f, 0.f, 0.f, 0.f);
      if (m < M)
        v = *reinterpret_cast<const float4*>(A + (size_t)m * lda + k0 + kk);
      const float vals[4] = {v.x, v.y, v.z, v.w};
      #pragma unroll
      for (int e = 0; e < 4; ++e) {
        const int k = kk + e;
        Asw[r >> 4][a_lane(r, k)][a_idx(k)] = (__bf16)vals[e];
      }
    }
    const char* pb = (const char*)(pw + ((size_t)kb * (N >> 5) + nb0) * 1024);
    async_b64(ldsB + t * 8, pb + t * 8);      // 2048 B fragment tile
    wait_async0();
    __syncthreads();

    const v16bf a  = *reinterpret_cast<const v16bf*>(&Asw[wm][lane][0]);
    const v16bf bb = *reinterpret_cast<const v16bf*>(&Bsw[wn][lane][0]);
    acc = wmma_bf16(a, bb, acc);
  }

  const int n     = nb0 * 32 + wn * 16 + (lane & 15);
  const int rowhi = (lane >> 4) * 8;
  const float bi  = bias[n];
  #pragma unroll
  for (int r = 0; r < 8; ++r) {
    const int m = m0 + wm * 16 + rowhi + r;
    if (m < M) {
      float v = acc[r] + bi;
      if (relu) v = fmaxf(v, 0.f);
      out[(size_t)m * N + n] = v;
    }
  }
}

// ---------------------------------------------------------------------------
// Final 1024 -> 3 projection (tiny; plain VALU).
// ---------------------------------------------------------------------------
__global__ void mlp_out_kernel(const float* __restrict__ h2,
                               const float* __restrict__ W,
                               const float* __restrict__ bias,
                               float* __restrict__ out)
{
  const int t = threadIdx.x;        // 0..383
  const int b = t / 3, c = t % 3;
  float s = bias[c];
  for (int k = 0; k < MLPDD; ++k) s += h2[b * MLPDD + k] * W[k * 3 + c];
  out[b * 3 + c] = s;
}

// ---------------------------------------------------------------------------
extern "C" void kernel_launch(void* const* d_in, const int* in_sizes, int n_in,
                              void* d_out, int out_size, void* d_ws, size_t ws_size,
                              hipStream_t stream) {
  const int*   sentences = (const int*)  d_in[0];
  const float* embed     = (const float*)d_in[2];
  const float* compW     = (const float*)d_in[3];
  const float* compb     = (const float*)d_in[4];
  const float* selW1     = (const float*)d_in[5];
  const float* selb1     = (const float*)d_in[6];
  const float* selW2     = (const float*)d_in[7];
  const float* selb2     = (const float*)d_in[8];
  const float* mlpW0     = (const float*)d_in[9];
  const float* mlpb0     = (const float*)d_in[10];
  const float* mlpW1     = (const float*)d_in[11];
  const float* mlpb1     = (const float*)d_in[12];
  const float* mlpWout   = (const float*)d_in[13];
  const float* mlpbout   = (const float*)d_in[14];

  float* ws = (float*)d_ws;
  const size_t SB = (size_t)NB * SSEQ * DST;     // 4,194,304 floats
  float* s0     = ws;
  float* s1     = s0 + SB;
  float* comp   = s1 + SB;
  float* logits = comp + SB;                     // NB*SSEQ
  float* h1     = logits + (size_t)NB * SSEQ;
  float* h2     = h1 + (size_t)NB * MLPDD;
  __bf16* sb0   = (__bf16*)(h2 + (size_t)NB * MLPDD);
  __bf16* sb1   = sb0 + SB;
  __bf16* pwC   = sb1 + SB;                      // 512*1280
  __bf16* pwS   = pwC + (size_t)DST * NGATE;     // 1024*128
  __bf16* pwM0  = pwS + (size_t)(2 * DST) * SELD;// 512*1024
  __bf16* pwM1  = pwM0 + (size_t)DST * MLPDD;    // 1024*1024

  // --- weight packing (fragment-ordered bf16) ---
  pack_compW_kernel<<<(DST * NGATE + 255) / 256, 256, 0, stream>>>(compW, pwC);
  pack_w_kernel<<<(2 * DST * SELD + 255) / 256, 256, 0, stream>>>(selW1, pwS, 2 * DST, SELD);
  pack_w_kernel<<<(DST * MLPDD + 255) / 256, 256, 0, stream>>>(mlpW0, pwM0, DST, MLPDD);
  pack_w_kernel<<<(MLPDD * MLPDD + 255) / 256, 256, 0, stream>>>(mlpW1, pwM1, MLPDD, MLPDD);

  embed_kernel<<<NB * SSEQ, 128, 0, stream>>>(sentences, embed, s0, sb0);

  float*  cur = s0;  float*  nxt = s1;
  __bf16* curb = sb0; __bf16* nxtb = sb1;
  for (int L = SSEQ - 1; L >= 1; --L) {
    const int M = NB * L;
    dim3 gg((M + 63) / 64, HH / 32);
    gates_kernel<<<gg, 256, 0, stream>>>(cur, curb, pwC, compb, comp, L);
    sel_kernel<<<(M + 15) / 16, 256, 0, stream>>>(curb, pwS, selb1, selW2, selb2,
                                                  logits, L);
    update_kernel<<<NB, 256, 0, stream>>>(cur, comp, logits, nxt, nxtb, L);
    float* tf = cur; cur = nxt; nxt = tf;
    __bf16* tb = curb; curb = nxtb; nxtb = tb;
  }

  mlp_gemm_kernel<<<dim3(2, MLPDD / 32), 256, 0, stream>>>(
      cur, SSEQ * DST, NB, DST, MLPDD, pwM0, mlpb0, h1, 1);
  mlp_gemm_kernel<<<dim3(2, MLPDD / 32), 256, 0, stream>>>(
      h1, MLPDD, NB, MLPDD, MLPDD, pwM1, mlpb1, h2, 1);
  mlp_out_kernel<<<1, 384, 0, stream>>>(h2, mlpWout, mlpbout, (float*)d_out);
}
